// GridAugmentedSelfAttention_37563783970981
// MI455X (gfx1250) — matrix operand order repaired and checked by reading, not compiled, not run on hardware
//
#include <hip/hip_runtime.h>

typedef unsigned short u16;
typedef __attribute__((ext_vector_type(16))) __bf16 v16bf;
typedef __attribute__((ext_vector_type(8)))  float  v8f;
typedef __attribute__((ext_vector_type(4)))  unsigned int u32x4;

// ---------- helpers ----------
__device__ __forceinline__ u16 f2bf(float x) {
  unsigned u = __builtin_bit_cast(unsigned, x);
  u += 0x7fffu + ((u >> 16) & 1u);          // round-to-nearest-even
  return (u16)(u >> 16);
}

union FragU {
  v16bf v;
  u32x4 q[2];
  u16   u[16];
};

// A-matrix 16x32 bf16 fragment (ISA 7.12.2): lane m=lane&15, hi=lane>>4
//   VGPR0-3: K = k0 + hi*8 .. +7 ; VGPR4-7: K = k0+16 + hi*8 .. +7
__device__ __forceinline__ v16bf ldA_bf(const u16* row, int k0, int hi) {
  FragU f;
  f.q[0] = *(const u32x4*)(row + k0 + hi * 8);
  f.q[1] = *(const u32x4*)(row + k0 + 16 + hi * 8);
  return f.v;
}
__device__ __forceinline__ v16bf ldA_f32(const float* row, int k0, int hi) {
  FragU f;
  const float* p0 = row + k0 + hi * 8;
  const float* p1 = row + k0 + 16 + hi * 8;
#pragma unroll
  for (int i = 0; i < 8; ++i) { f.u[i] = f2bf(p0[i]); f.u[8 + i] = f2bf(p1[i]); }
  return f.v;
}
// B-matrix 32x16 bf16 fragment: lane n=lane&15, same K order as A:
//   K = k0 + hi*8 .. +7, then k0+16 + hi*8 .. +7 from "row n holds K contiguous" storage.
__device__ __forceinline__ v16bf ldB_bf(const u16* row, int k0, int hi) {
  FragU f;
  f.q[0] = *(const u32x4*)(row + k0 + hi * 8);
  f.q[1] = *(const u32x4*)(row + k0 + 16 + hi * 8);
  return f.v;
}
__device__ __forceinline__ v16bf ldB_f32(const float* row, int k0, int hi) {
  FragU f;
  const float* p0 = row + k0 + hi * 8;
  const float* p1 = row + k0 + 16 + hi * 8;
#pragma unroll
  for (int i = 0; i < 8; ++i) { f.u[i] = f2bf(p0[i]); f.u[8 + i] = f2bf(p1[i]); }
  return f.v;
}

__device__ __forceinline__ v8f wmma_bf16(v16bf a, v16bf b, v8f c) {
  return __builtin_amdgcn_wmma_f32_16x16x32_bf16(false, a, false, b,
                                                 (short)0, c, false, false);
}

// ---------- kernels ----------
__global__ void cvt_kernel(const float* __restrict__ s, u16* __restrict__ d, int n) {
  int i = blockIdx.x * blockDim.x + threadIdx.x;
  if (i < n) d[i] = f2bf(s[i]);
}

// zero the t=144..159 pad columns of VT[b][d][160]
__global__ void vtpad_kernel(u16* __restrict__ vt) {
  int i = blockIdx.x * blockDim.x + threadIdx.x;   // 128*1024*16
  int b = i >> 14;
  int r = i & 16383;
  int d = r >> 4;
  int t = 144 + (r & 15);
  vt[(size_t)b * 163840 + d * 160 + t] = 0;
}

// rp2[s][t] = relu( sum_d relu(pos(s,t)·Wlin[d]) * w_g[s][d] )
__global__ void posbias_kernel(const float* __restrict__ Wlin,
                               const float* __restrict__ wg,
                               float* __restrict__ rp2) {
  int idx = blockIdx.x * 256 + threadIdx.x;        // 144*144 = 20736 exact
  int s = idx / 144, t = idx - s * 144;
  const float inv = 1.0f / 12.0f;
  float cxs = ((s % 12) + 0.5f) * inv, cys = ((s / 12) + 0.5f) * inv;
  float cxt = ((t % 12) + 0.5f) * inv, cyt = ((t / 12) + 0.5f) * inv;
  float dx = cxs - cxt, dy = cys - cyt;
  float ax = (dx > 0.f) ? 0.5f : ((dx < 0.f) ? -0.5f : 0.f);
  float ay = (dy > 0.f) ? 0.5f : ((dy < 0.f) ? -0.5f : 0.f);
  float fx = fabsf(__logf(fabsf(dx) + 0.5f));
  float fy = fabsf(__logf(fabsf(dy) + 0.5f));
  const float* wgr = wg + s * 1024;
  float acc = 0.f;
  for (int d = 0; d < 1024; ++d) {
    const float4 w = *(const float4*)(Wlin + d * 4);
    float h = fx * w.x + fy * w.y + ax * w.z + ay * w.w;
    h = fmaxf(h, 0.f);
    acc = fmaf(h, wgr[d], acc);
  }
  rp2[idx] = fmaxf(acc, 0.f);
}

// MODE 0: out[m][n] = relu(X[m]·W[n] + bias[n]), X f32 (Mx1024), W bf16 (1024x1024)
//         M = 18432 tokens, N = 1024 features, normal bf16 store.
// MODE 1: same math with A = W rows (d), B = X rows (tokens):
//         out'[d][tok] -> VT[b][d][t] transposed bf16 store, bias on d.
template <int MODE>
__global__ __launch_bounds__(256) void proj_kernel(const float* __restrict__ X,
                                                   const u16* __restrict__ W,
                                                   const float* __restrict__ bias,
                                                   u16* __restrict__ out) {
  const int tid = threadIdx.x;
  const int lane = tid & 31;
  const int wid = tid >> 5;
  const int ln = lane & 15, hi = lane >> 4;
  const int wm = wid & 3, wn = wid >> 2;
  const int mbase = blockIdx.x * 128 + wm * 32;    // 2 m-frags
  const int nbase = blockIdx.y * 128 + wn * 64;    // 4 n-frags

  v8f acc[2][4] = {};

  const u16*   aW = W + (size_t)(mbase + ln) * 1024;       // MODE1 A rows
  const float* aX = X + (size_t)(mbase + ln) * 1024;       // MODE0 A rows

  for (int k0 = 0; k0 < 1024; k0 += 32) {
    // prefetch the streamed rows ~4 k-steps (512B f32 / 256B bf16) ahead
    if (k0 + 128 < 1024) {
#pragma unroll
      for (int fm = 0; fm < 2; ++fm) {
        if (MODE == 0) __builtin_prefetch(aX + fm * 16 * 1024 + k0 + 128, 0, 3);
        else           __builtin_prefetch(aW + fm * 16 * 1024 + k0 + 128, 0, 3);
      }
    }
    v16bf a[2];
#pragma unroll
    for (int fm = 0; fm < 2; ++fm) {
      if (MODE == 0) a[fm] = ldA_f32(aX + fm * 16 * 1024, k0, hi);
      else           a[fm] = ldA_bf (aW + fm * 16 * 1024, k0, hi);
    }
    v16bf b[4];
#pragma unroll
    for (int fn = 0; fn < 4; ++fn) {
      if (MODE == 0) b[fn] = ldB_bf (W + (size_t)(nbase + fn * 16 + ln) * 1024, k0, hi);
      else           b[fn] = ldB_f32(X + (size_t)(nbase + fn * 16 + ln) * 1024, k0, hi);
    }
#pragma unroll
    for (int fn = 0; fn < 4; ++fn)
#pragma unroll
      for (int fm = 0; fm < 2; ++fm) acc[fm][fn] = wmma_bf16(a[fm], b[fn], acc[fm][fn]);
  }

#pragma unroll
  for (int fm = 0; fm < 2; ++fm)
#pragma unroll
    for (int fn = 0; fn < 4; ++fn)
#pragma unroll
      for (int j = 0; j < 8; ++j) {
        int mrow = mbase + fm * 16 + j + hi * 8;
        int ncol = nbase + fn * 16 + ln;
        float v = acc[fm][fn][j];
        if (MODE == 0) {
          v = fmaxf(v + bias[ncol], 0.f);
          out[(size_t)mrow * 1024 + ncol] = f2bf(v);
        } else {
          v = fmaxf(v + bias[mrow], 0.f);
          int b = ncol / 144, t = ncol - b * 144;
          out[(size_t)b * 163840 + mrow * 160 + t] = f2bf(v);
        }
      }
}

// One wave per (b, 16-query tile): scores -> bias -> softmax -> PV.
__global__ __launch_bounds__(256) void attn_kernel(const u16* __restrict__ Q,
                                                   const u16* __restrict__ K,
                                                   const u16* __restrict__ VT,
                                                   const float* __restrict__ rp2,
                                                   float* __restrict__ out) {
  __shared__ __align__(16) u16 P[8 * 16 * 160];    // per-wave 16x160 bf16 probs

  const int tid = threadIdx.x;
  const int lane = tid & 31, wid = tid >> 5;
  const int ln = lane & 15, hi = lane >> 4;
  const int task = blockIdx.x * 8 + wid;           // 144*8 = 1152 = 128*9
  const int b = task / 9, mt = task - b * 9;
  const int m0 = mt * 16;

  const u16* Qb = Q + (size_t)b * 144 * 1024;
  const u16* Kb = K + (size_t)b * 144 * 1024;
  const u16* Vb = VT + (size_t)b * 163840;

  // ---- scores: 16 x 144 ----
  v8f acc[9] = {};
  const u16* qrow = Qb + (size_t)(m0 + ln) * 1024;
  const u16* krow = Kb + (size_t)ln * 1024;        // row ln of each 16-row K tile
  for (int k0 = 0; k0 < 1024; k0 += 32) {
    if (k0 + 128 < 1024) {
      __builtin_prefetch(qrow + k0 + 128, 0, 3);
#pragma unroll
      for (int nt = 0; nt < 9; ++nt)
        __builtin_prefetch(krow + (size_t)nt * 16 * 1024 + k0 + 128, 0, 3);
    }
    v16bf a = ldA_bf(qrow, k0, hi);
    v16bf bk[9];
#pragma unroll
    for (int nt = 0; nt < 9; ++nt)
      bk[nt] = ldB_bf(krow + (size_t)nt * 16 * 1024, k0, hi);
#pragma unroll
    for (int nt = 0; nt < 9; ++nt) acc[nt] = wmma_bf16(a, bk[nt], acc[nt]);
  }

  // ---- bias + softmax (rows live across 16-lane halves) ----
  u16* Pw = P + wid * 2560;
#pragma unroll
  for (int j = 0; j < 8; ++j) {
    const int srow = m0 + j + hi * 8;
    const float* rpr = rp2 + srow * 144 + ln;
    float mx = -3.0e38f;
#pragma unroll
    for (int nt = 0; nt < 9; ++nt) {
      float v = acc[nt][j] * 0.03125f + rpr[nt * 16];
      acc[nt][j] = v;
      mx = fmaxf(mx, v);
    }
    mx = fmaxf(mx, __shfl_xor(mx, 1, 32));
    mx = fmaxf(mx, __shfl_xor(mx, 2, 32));
    mx = fmaxf(mx, __shfl_xor(mx, 4, 32));
    mx = fmaxf(mx, __shfl_xor(mx, 8, 32));
    float sum = 0.f;
#pragma unroll
    for (int nt = 0; nt < 9; ++nt) {
      float e = __expf(acc[nt][j] - mx);
      acc[nt][j] = e;
      sum += e;
    }
    sum += __shfl_xor(sum, 1, 32);
    sum += __shfl_xor(sum, 2, 32);
    sum += __shfl_xor(sum, 4, 32);
    sum += __shfl_xor(sum, 8, 32);
    float rinv = 1.0f / sum;
#pragma unroll
    for (int nt = 0; nt < 9; ++nt)
      Pw[(j + hi * 8) * 160 + nt * 16 + ln] = f2bf(acc[nt][j] * rinv);
  }
#pragma unroll
  for (int j = 0; j < 8; ++j)                       // zero K-pad 144..159
    Pw[(j + hi * 8) * 160 + 144 + ln] = 0;
  __syncthreads();

  // ---- PV: out 16 x 1024, K-dim padded to 160 ----
  v16bf Ap[5];
  const u16* prow = Pw + ln * 160;
#pragma unroll
  for (int kt = 0; kt < 5; ++kt) Ap[kt] = ldA_bf(prow, kt * 32, hi);

  const u16* vrow = Vb + (size_t)ln * 160;
  for (int dc = 0; dc < 16; ++dc) {
    const int d0 = dc * 64;
    v8f o[4] = {};
#pragma unroll
    for (int kt = 0; kt < 5; ++kt) {
      v16bf bf[4];
#pragma unroll
      for (int fn = 0; fn < 4; ++fn)
        bf[fn] = ldB_bf(vrow + (size_t)(d0 + fn * 16) * 160, kt * 32, hi);
#pragma unroll
      for (int fn = 0; fn < 4; ++fn) o[fn] = wmma_bf16(Ap[kt], bf[fn], o[fn]);
    }
#pragma unroll
    for (int fn = 0; fn < 4; ++fn)
#pragma unroll
      for (int j = 0; j < 8; ++j) {
        int srow = m0 + j + hi * 8;
        out[((size_t)(b * 144 + srow)) * 1024 + d0 + fn * 16 + ln] = o[fn][j];
      }
  }
}

// ---------- launch ----------
extern "C" void kernel_launch(void* const* d_in, const int* in_sizes, int n_in,
                              void* d_out, int out_size, void* d_ws, size_t ws_size,
                              hipStream_t stream) {
  const float* query = (const float*)d_in[0];
  const float* key   = (const float*)d_in[1];
  const float* value = (const float*)d_in[2];
  const float* Wq    = (const float*)d_in[3];
  const float* bq    = (const float*)d_in[4];
  const float* Wk    = (const float*)d_in[5];
  const float* bk    = (const float*)d_in[6];
  const float* Wv    = (const float*)d_in[7];
  const float* bv    = (const float*)d_in[8];
  const float* Wlin  = (const float*)d_in[9];
  const float* w_g   = (const float*)d_in[10];

  // workspace layout (u16 elements)
  u16* WB  = (u16*)d_ws;                    // 3 x 1024*1024 bf16 weights
  u16* Qb  = WB + 3u * 1048576u;            // 18432*1024 bf16
  u16* Kb  = Qb + 18874368u;                // 18432*1024 bf16
  u16* VT  = Kb + 18874368u;                // 128*1024*160 bf16 (transposed V)
  float* rp2 = (float*)(VT + 20971520u);    // 144*144 f32

  cvt_kernel<<<4096, 256, 0, stream>>>(Wq, WB,             1048576);
  cvt_kernel<<<4096, 256, 0, stream>>>(Wk, WB + 1048576u,  1048576);
  cvt_kernel<<<4096, 256, 0, stream>>>(Wv, WB + 2097152u,  1048576);
  vtpad_kernel<<<8192, 256, 0, stream>>>(VT);
  posbias_kernel<<<81, 256, 0, stream>>>(Wlin, w_g, rp2);

  proj_kernel<0><<<dim3(144, 8), 256, 0, stream>>>(query, WB,             bq, Qb);
  proj_kernel<0><<<dim3(144, 8), 256, 0, stream>>>(key,   WB + 1048576u,  bk, Kb);
  proj_kernel<1><<<dim3(8, 144), 256, 0, stream>>>(value, WB + 2097152u,  bv, VT);

  attn_kernel<<<144, 256, 0, stream>>>(Qb, Kb, VT, rp2, (float*)d_out);
}
